// GNN_18957985644609
// MI455X (gfx1250) — matrix operand order repaired and checked by reading, not compiled, not run on hardware
//
#include <hip/hip_runtime.h>
#include <hip/hip_bf16.h>

typedef __attribute__((ext_vector_type(16))) _Float16 v16h;
typedef __attribute__((ext_vector_type(8)))  _Float16 v8h;
typedef __attribute__((ext_vector_type(8)))  float    v8f;
typedef int v4i_g __attribute__((vector_size(16)));   // matches builtin param type

#define GN_N   50000
#define GN_E   200000
#define GN_G   512
#define GN_FN  64
#define GN_FE  16
#define GN_H   32
#define GN_L   3
#define GN_EPS 1e-5f
#define NTILES (GN_E / 16)   // 12500 exactly
#define WCH    33            // 32 ea-feature chunks + 1 bias chunk

#if __has_builtin(__builtin_amdgcn_global_load_async_to_lds_b128) && \
    __has_builtin(__builtin_amdgcn_s_wait_asynccnt)
#define GN_ASYNC_STAGE 1
#endif

// ---------------------------------------------------------------------------
// Node embedding: h = x @ W_node + b_node  (f32 + f16 copy for WMMA A-operand)
// ---------------------------------------------------------------------------
__global__ void k_node_emb(const float* __restrict__ x, const float* __restrict__ Wn,
                           const float* __restrict__ bn, float* __restrict__ h,
                           _Float16* __restrict__ h16)
{
    int idx = blockIdx.x * blockDim.x + threadIdx.x;
    if (idx >= GN_N * GN_H) return;
    int n = idx >> 5, c = idx & 31;
    float acc = bn[c];
    const float* xr = x + (size_t)n * GN_FN;
    #pragma unroll 8
    for (int k = 0; k < GN_FN; ++k) acc = fmaf(xr[k], Wn[k * GN_H + c], acc);
    h[idx] = acc;
    h16[idx] = (_Float16)acc;
}

// ---------------------------------------------------------------------------
// Edge embedding: ea16 = f16(edge_attr @ W_edge + b_edge)   [E, 32]
// ---------------------------------------------------------------------------
__global__ void k_edge_emb(const float* __restrict__ eattr, const float* __restrict__ We,
                           const float* __restrict__ be, _Float16* __restrict__ ea16)
{
    int idx = blockIdx.x * blockDim.x + threadIdx.x;
    if (idx >= GN_E * GN_H) return;
    int e = idx >> 5, c = idx & 31;
    float acc = be[c];
    const float* er = eattr + (size_t)e * GN_FE;
    #pragma unroll
    for (int k = 0; k < GN_FE; ++k) acc = fmaf(er[k], We[k * GN_H + c], acc);
    ea16[idx] = (_Float16)acc;
}

// ---------------------------------------------------------------------------
// Re-pack W_nn[l] (32x1024 f32) + b_nn[l] (1024) into this kernel's contiguous-k WMMA B-operand
// lane layout, f16:  wprep[kk][ntile][lane][pos],  pos=2v+hb,
//   j = pos + (lane>=16 ? 16 : 0)   (WMMA-K index, == h-feature i)
//   o = ntile*16 + (lane&15)        (output column)
//   value = kk<32 ? W_nn[kk, j*32+o] : b_nn[j*32+o]
// ---------------------------------------------------------------------------
__global__ void k_prep_w(const float* __restrict__ Wnn, const float* __restrict__ bnn,
                         _Float16* __restrict__ wprep)
{
    int i = blockIdx.x * blockDim.x + threadIdx.x;
    if (i >= WCH * 2 * 32 * 16) return;          // 33792
    int kk  = i >> 10;                           // 2*32*16 = 1024
    int rem = i & 1023;
    int t   = rem >> 9;
    int r2  = rem & 511;
    int L   = r2 >> 4;
    int pos = r2 & 15;
    int j   = pos + ((L >= 16) ? 16 : 0);
    int o   = t * 16 + (L & 15);
    float v = (kk < 32) ? Wnn[kk * 1024 + j * 32 + o] : bnn[j * 32 + o];
    wprep[i] = (_Float16)v;
}

// ---------------------------------------------------------------------------
// Fused NNConv message + scatter:  agg[dst[e]] += h16[src[e]] @ (ea@W_nn+b_nn)
// as GEMM  M=E (tile 16), K=1056, N=32 with on-the-fly rank-1 A chunks.
// One wave per 16-edge tile; 66 v_wmma_f32_16x16x32_f16 per tile.
// kk-loop kept rolled (unroll 2) so only one/two B tiles are live: no spills,
// latency hidden by 8 waves/SIMD of occupancy.
// ---------------------------------------------------------------------------
__launch_bounds__(256)
__global__ void k_edge_msg(const _Float16* __restrict__ h16,
                           const _Float16* __restrict__ ea16,
                           const _Float16* __restrict__ wprep,
                           const int* __restrict__ src,
                           const int* __restrict__ dstv,
                           float* __restrict__ agg)
{
    __shared__ __align__(16) _Float16 wsh[WCH * 2 * 32 * 16];   // 67.6 KB (CDNA5 320KB LDS)
    __shared__ __align__(16) _Float16 eash[8][16 * GN_H];       // per-wave ea tile
    __shared__ int dsh[8][16];                                  // per-wave dst tile

    // Stage packed B into LDS once per workgroup.
    {
        const uint4* g4 = (const uint4*)wprep;
        uint4* l4 = (uint4*)wsh;
#ifdef GN_ASYNC_STAGE
        // gfx1250 async copy: global -> LDS without VGPR round trip (ASYNCcnt).
        for (int i = threadIdx.x; i < (WCH * 2 * 32 * 16) / 8; i += blockDim.x) {
            __builtin_amdgcn_global_load_async_to_lds_b128(
                (v4i_g*)(g4 + i), (v4i_g*)(l4 + i), 0, 0);
        }
        __builtin_amdgcn_s_wait_asynccnt(0);
#else
        for (int i = threadIdx.x; i < (WCH * 2 * 32 * 16) / 8; i += blockDim.x) l4[i] = g4[i];
#endif
    }
    __syncthreads();

    const int lane = threadIdx.x & 31;
    const int wave = threadIdx.x >> 5;
    const int half = lane >> 4;        // 0: lanes 0-15, 1: lanes 16-31
    const int m    = lane & 15;        // edge row within tile / output column
    const int wid  = blockIdx.x * (blockDim.x >> 5) + wave;
    const int nw   = gridDim.x * (blockDim.x >> 5);
    const _Float16* wl = wsh + lane * 16;   // this lane's B fragment base

    for (int tile = wid; tile < NTILES; tile += nw) {
        const int e  = tile * 16 + m;              // E is a multiple of 16
        const int sn = src[e];

        // Gathered h row, contiguous k order (matches wprep B):
        // lane<16 holds j {0..15}; lane>=16 holds j {16..31}.
        const _Float16* hrow = h16 + (size_t)sn * GN_H;
        v8h hlo = *(const v8h*)(hrow + half * 16);
        v8h hhi = *(const v8h*)(hrow + half * 16 + 8);
        v16h hreg;
        #pragma unroll
        for (int i = 0; i < 8; ++i) { hreg[i] = hlo[i]; hreg[i + 8] = hhi[i]; }

        // Stage this tile's ea rows and dst indices (intra-wave LDS, DS ops in-order).
        const _Float16* earow = ea16 + (size_t)e * GN_H;
        *(v8h*)&eash[wave][m * GN_H + half * 16] = *(const v8h*)(earow + half * 16);
        if (half == 0) dsh[wave][m] = dstv[e];

        v8f c0 = {}; v8f c1 = {};
        #pragma unroll 2
        for (int kk = 0; kk < 32; ++kk) {
            _Float16 s = eash[wave][m * GN_H + kk];        // ea[e_m, kk] broadcast scale
            v16h a  = hreg * s;                            // rank-1 A chunk (v_pk_mul_f16)
            v16h b0 = *(const v16h*)(wl + kk * 1024);
            v16h b1 = *(const v16h*)(wl + kk * 1024 + 512);
            c0 = __builtin_amdgcn_wmma_f32_16x16x32_f16(false, a, false, b0, (short)0, c0, false, false);
            c1 = __builtin_amdgcn_wmma_f32_16x16x32_f16(false, a, false, b1, (short)0, c1, false, false);
        }
        {   // bias chunk: A = h row, B = b_nn reshaped
            v16h b0 = *(const v16h*)(wl + 32 * 1024);
            v16h b1 = *(const v16h*)(wl + 32 * 1024 + 512);
            c0 = __builtin_amdgcn_wmma_f32_16x16x32_f16(false, hreg, false, b0, (short)0, c0, false, false);
            c1 = __builtin_amdgcn_wmma_f32_16x16x32_f16(false, hreg, false, b1, (short)0, c1, false, false);
        }

        // C/D layout: VGPR r -> edge row (r + half*8), lane&15 -> column.
        #pragma unroll
        for (int r = 0; r < 8; ++r) {
            const int em = r + half * 8;
            const int dn = dsh[wave][em];
            float* base = agg + (size_t)dn * GN_H + m;
            unsafeAtomicAdd(base,      c0[r]);
            unsafeAtomicAdd(base + 16, c1[r]);
        }
    }
}

// ---------------------------------------------------------------------------
// hpre = h @ W_root[l] + agg + b_conv[l]; accumulate per-column sum/sumsq
// ---------------------------------------------------------------------------
__launch_bounds__(256)
__global__ void k_update(const float* __restrict__ h, const float* __restrict__ agg,
                         const float* __restrict__ Wr, const float* __restrict__ bc,
                         float* __restrict__ hpre, float* __restrict__ stats)
{
    __shared__ float wsh[GN_H * GN_H];
    __shared__ float ssum[GN_H];
    __shared__ float ssq[GN_H];
    if (threadIdx.x < GN_H) { ssum[threadIdx.x] = 0.f; ssq[threadIdx.x] = 0.f; }
    for (int i = threadIdx.x; i < GN_H * GN_H; i += blockDim.x) wsh[i] = Wr[i];
    __syncthreads();

    int n = blockIdx.x * blockDim.x + threadIdx.x;
    if (n < GN_N) {
        float row[GN_H];
        #pragma unroll
        for (int i = 0; i < GN_H; ++i) row[i] = h[(size_t)n * GN_H + i];
        #pragma unroll
        for (int o = 0; o < GN_H; ++o) {
            float acc = bc[o] + agg[(size_t)n * GN_H + o];
            #pragma unroll
            for (int i = 0; i < GN_H; ++i) acc = fmaf(row[i], wsh[i * GN_H + o], acc);
            hpre[(size_t)n * GN_H + o] = acc;
            atomicAdd(&ssum[o], acc);
            atomicAdd(&ssq[o], acc * acc);
        }
    }
    __syncthreads();
    if (threadIdx.x < GN_H) {
        unsafeAtomicAdd(&stats[threadIdx.x], ssum[threadIdx.x]);
        unsafeAtomicAdd(&stats[GN_H + threadIdx.x], ssq[threadIdx.x]);
    }
}

// ---------------------------------------------------------------------------
// BatchNorm + ReLU; writes f32 h and f16 copy for next layer's WMMA
// ---------------------------------------------------------------------------
__global__ void k_bn(const float* __restrict__ hpre, const float* __restrict__ stats,
                     const float* __restrict__ gam, const float* __restrict__ bet,
                     float* __restrict__ h, _Float16* __restrict__ h16)
{
    int idx = blockIdx.x * blockDim.x + threadIdx.x;
    if (idx >= GN_N * GN_H) return;
    int c = idx & (GN_H - 1);
    const float inv = 1.0f / (float)GN_N;
    float mu  = stats[c] * inv;
    float var = stats[GN_H + c] * inv - mu * mu;
    float v = (hpre[idx] - mu) * rsqrtf(var + GN_EPS) * gam[c] + bet[c];
    v = v > 0.f ? v : 0.f;
    h[idx] = v;
    h16[idx] = (_Float16)v;
}

__global__ void k_zero(float* __restrict__ p, int n)
{
    int i = blockIdx.x * blockDim.x + threadIdx.x;
    if (i < n) p[i] = 0.f;
}

// ---------------------------------------------------------------------------
// Global mean pool accumulation (batch is sorted, atomics over G=512 graphs)
// ---------------------------------------------------------------------------
__global__ void k_pool(const float* __restrict__ h, const int* __restrict__ batch,
                       float* __restrict__ gsum, float* __restrict__ cnt)
{
    int idx = blockIdx.x * blockDim.x + threadIdx.x;
    if (idx >= GN_N * GN_H) return;
    int n = idx >> 5, c = idx & 31;
    int g = batch[n];
    unsafeAtomicAdd(&gsum[(size_t)g * GN_H + c], h[idx]);
    if (c == 0) unsafeAtomicAdd(&cnt[g], 1.0f);
}

// ---------------------------------------------------------------------------
// Head: gx = gsum/max(cnt,1); relu(gx@W2+b2)@W3+b3
// ---------------------------------------------------------------------------
__global__ void k_head(const float* __restrict__ gsum, const float* __restrict__ cnt,
                       const float* __restrict__ W2, const float* __restrict__ b2,
                       const float* __restrict__ W3, const float* __restrict__ b3,
                       float* __restrict__ out)
{
    int g = blockIdx.x * blockDim.x + threadIdx.x;
    if (g >= GN_G) return;
    float ct = cnt[g]; ct = ct > 1.0f ? ct : 1.0f;
    float inv = 1.0f / ct;
    float gx[GN_H];
    #pragma unroll
    for (int i = 0; i < GN_H; ++i) gx[i] = gsum[(size_t)g * GN_H + i] * inv;
    float acc = b3[0];
    #pragma unroll
    for (int j = 0; j < GN_H / 2; ++j) {
        float hh = b2[j];
        #pragma unroll
        for (int i = 0; i < GN_H; ++i) hh = fmaf(gx[i], W2[i * (GN_H / 2) + j], hh);
        hh = hh > 0.f ? hh : 0.f;
        acc = fmaf(hh, W3[j], acc);
    }
    out[g] = acc;
}

// ---------------------------------------------------------------------------
extern "C" void kernel_launch(void* const* d_in, const int* in_sizes, int n_in,
                              void* d_out, int out_size, void* d_ws, size_t ws_size,
                              hipStream_t stream)
{
    const float* x      = (const float*)d_in[0];
    const float* eattr  = (const float*)d_in[1];
    const float* W_node = (const float*)d_in[2];
    const float* b_node = (const float*)d_in[3];
    const float* W_edge = (const float*)d_in[4];
    const float* b_edge = (const float*)d_in[5];
    const float* W_nn   = (const float*)d_in[6];
    const float* b_nn   = (const float*)d_in[7];
    const float* W_root = (const float*)d_in[8];
    const float* b_conv = (const float*)d_in[9];
    const float* gam    = (const float*)d_in[10];
    const float* bet    = (const float*)d_in[11];
    const float* W2     = (const float*)d_in[12];
    const float* b2     = (const float*)d_in[13];
    const float* W3     = (const float*)d_in[14];
    const float* b3     = (const float*)d_in[15];
    const int*   eidx   = (const int*)d_in[16];
    const int*   batch  = (const int*)d_in[17];
    const int* srcI = eidx;            // edge_index[0]
    const int* dstI = eidx + GN_E;     // edge_index[1]

    char* ws = (char*)d_ws;
    auto alloc = [&](size_t bytes) -> char* {
        char* p = ws; ws += (bytes + 255) & ~(size_t)255; return p;
    };
    float*    h     = (float*)   alloc(sizeof(float)    * GN_N * GN_H);
    float*    hpre  = (float*)   alloc(sizeof(float)    * GN_N * GN_H);
    float*    agg   = (float*)   alloc(sizeof(float)    * GN_N * GN_H);
    _Float16* h16   = (_Float16*)alloc(sizeof(_Float16) * GN_N * GN_H);
    _Float16* ea16  = (_Float16*)alloc(sizeof(_Float16) * GN_E * GN_H);
    _Float16* wprep = (_Float16*)alloc(sizeof(_Float16) * WCH * 2 * 32 * 16);
    float*    stats = (float*)   alloc(sizeof(float)    * 2 * GN_H);
    float*    gsum  = (float*)   alloc(sizeof(float)    * GN_G * GN_H);
    float*    cnt   = (float*)   alloc(sizeof(float)    * GN_G);

    const int NT = 256;
    k_node_emb<<<(GN_N * GN_H + NT - 1) / NT, NT, 0, stream>>>(x, W_node, b_node, h, h16);
    k_edge_emb<<<(GN_E * GN_H + NT - 1) / NT, NT, 0, stream>>>(eattr, W_edge, b_edge, ea16);

    for (int l = 0; l < GN_L; ++l) {
        k_prep_w<<<(WCH * 2 * 32 * 16 + NT - 1) / NT, NT, 0, stream>>>(
            W_nn + (size_t)l * 32 * 1024, b_nn + (size_t)l * 1024, wprep);
        k_zero<<<(GN_N * GN_H + NT - 1) / NT, NT, 0, stream>>>(agg, GN_N * GN_H);
        k_zero<<<1, 64, 0, stream>>>(stats, 2 * GN_H);
        k_edge_msg<<<512, 256, 0, stream>>>(h16, ea16, wprep, srcI, dstI, agg);
        k_update<<<(GN_N + NT - 1) / NT, NT, 0, stream>>>(
            h, agg, W_root + (size_t)l * GN_H * GN_H, b_conv + (size_t)l * GN_H, hpre, stats);
        k_bn<<<(GN_N * GN_H + NT - 1) / NT, NT, 0, stream>>>(
            hpre, stats, gam + l * GN_H, bet + l * GN_H, h, h16);
    }

    k_zero<<<(GN_G * GN_H + NT - 1) / NT, NT, 0, stream>>>(gsum, GN_G * GN_H);
    k_zero<<<(GN_G + NT - 1) / NT, NT, 0, stream>>>(cnt, GN_G);
    k_pool<<<(GN_N * GN_H + NT - 1) / NT, NT, 0, stream>>>(h, batch, gsum, cnt);
    k_head<<<(GN_G + NT - 1) / NT, NT, 0, stream>>>(gsum, cnt, W2, b2, W3, b3, (float*)d_out);
}
